// TransformerDecoderLayer_16363825398311
// MI455X (gfx1250) — matrix operand order repaired and checked by reading, not compiled, not run on hardware
//
#include <hip/hip_runtime.h>
#include <math.h>

// ---------------------------------------------------------------------------
// Types for CDNA5 WMMA (wave32): A/B = 16 bf16 (8 VGPRs), C/D = 8 f32.
// ---------------------------------------------------------------------------
typedef __attribute__((ext_vector_type(16))) __bf16 v16bf;
typedef __attribute__((ext_vector_type(8)))  float  v8f;
typedef __attribute__((ext_vector_type(4)))  unsigned int v4u;
typedef __attribute__((ext_vector_type(4)))  int v4i_t;

union Frag { v16bf v; v4u u[2]; };

__device__ __forceinline__ __bf16 f2bf(float f) {
    union { unsigned u; float f; } uf; uf.f = f;
    unsigned r = (uf.u + 0x7FFFu + ((uf.u >> 16) & 1u)) >> 16;  // RNE
    union { unsigned short s; __bf16 b; } ub; ub.s = (unsigned short)r;
    return ub.b;
}

__device__ __forceinline__ v8f wmma_bf16(const Frag& a, const Frag& b, v8f c) {
    return __builtin_amdgcn_wmma_f32_16x16x32_bf16(
        false, a.v, false, b.v, (short)0, c, false, false);
}

// ---------------------------------------------------------------------------
// CDNA5 async global->LDS copy (16B per lane), tracked by ASYNCcnt.
// Builtin signature (from probe diagnostic): (AS1 v4i* src, AS3 v4i* dst, imm, imm)
// ---------------------------------------------------------------------------
__device__ __forceinline__ void async_cp16(const __bf16* g, __bf16* l) {
#if __has_builtin(__builtin_amdgcn_global_load_async_to_lds_b128)
    __builtin_amdgcn_global_load_async_to_lds_b128(
        (__attribute__((address_space(1))) v4i_t*)(g),
        (__attribute__((address_space(3))) v4i_t*)(l),
        0, 0);
#else
    unsigned off = (unsigned)(unsigned long long)(uintptr_t)l;  // low 32b of generic = LDS offset
    asm volatile("global_load_async_to_lds_b128 %0, %1, off"
                 :: "v"(off), "v"((unsigned long long)(uintptr_t)g) : "memory");
#endif
}

#if __has_builtin(__builtin_amdgcn_s_wait_asynccnt)
#define WAIT_ASYNC(n) __builtin_amdgcn_s_wait_asynccnt(n)
#else
#define WAIT_ASYNC(n) asm volatile("s_wait_asynccnt %0" :: "i"(n) : "memory")
#endif

#define NEG_BIG 3.0e38f

// ---------------------------------------------------------------------------
// Weight repacking: fp32 -> bf16, N-major ("Bt") layout  Bt[n*K + k].
// ---------------------------------------------------------------------------
__global__ __launch_bounds__(256) void pack_headed(const float* __restrict__ W,
                                                   __bf16* __restrict__ out) {
    int i = blockIdx.x * 256 + threadIdx.x;      // i = n*1024 + k, 1M total
    int n = i >> 10, k = i & 1023;
    out[i] = f2bf(W[(size_t)(n >> 6) * 65536 + k * 64 + (n & 63)]);
}

__global__ __launch_bounds__(256) void pack_T(const float* __restrict__ W,
                                              __bf16* __restrict__ out,
                                              int N, int K) {
    int i = blockIdx.x * 256 + threadIdx.x;
    if (i >= N * K) return;
    int n = i / K, k = i - n * K;
    out[i] = f2bf(W[(size_t)k * N + n]);
}

// ---------------------------------------------------------------------------
// LayerNorm over rows of 1024, fp32 in -> bf16 out. One block (256 thr) / row.
// ---------------------------------------------------------------------------
__global__ __launch_bounds__(256) void ln_kernel(const float* __restrict__ x,
                                                 const float* __restrict__ w,
                                                 const float* __restrict__ b,
                                                 __bf16* __restrict__ out) {
    __shared__ float s1[8], s2[8];
    int row = blockIdx.x;
    const float* xr = x + (size_t)row * 1024;
    float4 v = *(const float4*)(xr + threadIdx.x * 4);
    float sum = v.x + v.y + v.z + v.w;
    float sq  = v.x * v.x + v.y * v.y + v.z * v.z + v.w * v.w;
#pragma unroll
    for (int off = 1; off < 32; off <<= 1) {
        sum += __shfl_xor(sum, off, 32);
        sq  += __shfl_xor(sq,  off, 32);
    }
    int wave = threadIdx.x >> 5, lane = threadIdx.x & 31;
    if (lane == 0) { s1[wave] = sum; s2[wave] = sq; }
    __syncthreads();
    float ts = 0.f, tq = 0.f;
#pragma unroll
    for (int i = 0; i < 8; i++) { ts += s1[i]; tq += s2[i]; }
    float mu   = ts * (1.0f / 1024.0f);
    float var  = tq * (1.0f / 1024.0f) - mu * mu;
    float rinv = rsqrtf(var + 1e-5f);
    int c = threadIdx.x * 4;
    __bf16* orow = out + (size_t)row * 1024;
    orow[c + 0] = f2bf((v.x - mu) * rinv * w[c + 0] + b[c + 0]);
    orow[c + 1] = f2bf((v.y - mu) * rinv * w[c + 1] + b[c + 1]);
    orow[c + 2] = f2bf((v.z - mu) * rinv * w[c + 2] + b[c + 2]);
    orow[c + 3] = f2bf((v.w - mu) * rinv * w[c + 3] + b[c + 3]);
}

// ---------------------------------------------------------------------------
// bf16 WMMA GEMM with async-to-LDS double buffering.
//   C[M,N] = A[M,K] (row-major) x Bt[N,K] (N-major).
// Block = 256 thr = 8 waves; block tile 128x128; wave tile 32(M) x 64(N).
// LDS rows padded to 40 bf16 (80B) -> conflict-free ds_load_b128.
// Epilogue modes:
//   0: bf16 [B,H,S,Dh] (Q/K)   1: bf16 [B,H,Dh,S] (V^T)
//   2: outf = aux[m,n] + c     3: outb = gelu(c + aux[n])
//   4: outf = c + aux[n] + aux2[m,n]
// ---------------------------------------------------------------------------
__device__ __forceinline__ void gemm_stage(const __bf16* __restrict__ A,
                                           const __bf16* __restrict__ Bt,
                                           int K, int kb, int mBlk, int nBlk,
                                           __bf16* sa, __bf16* sb, int tid) {
    const __bf16* gA = A  + (size_t)mBlk * K + kb;
    const __bf16* gB = Bt + (size_t)nBlk * K + kb;
#pragma unroll
    for (int i = 0; i < 2; i++) {                 // 512 16B-chunks per matrix / 256 thr
        int c = tid + i * 256;
        int row = c >> 2, off = (c & 3) * 8;      // 128 rows x 32 cols
        async_cp16(gA + (size_t)row * K + off, sa + row * 40 + off);
        async_cp16(gB + (size_t)row * K + off, sb + row * 40 + off);
    }
}

__global__ __launch_bounds__(256) void gemm_bf16_kernel(
    const __bf16* __restrict__ A, const __bf16* __restrict__ Bt,
    int N, int K, int mode,
    const float* __restrict__ aux, const float* __restrict__ aux2,
    float* __restrict__ outf, __bf16* __restrict__ outb) {
    __shared__ __bf16 sA[2][128 * 40];
    __shared__ __bf16 sB[2][128 * 40];
    const int tid  = threadIdx.x;
    const int lane = tid & 31;
    const int wave = tid >> 5;
    const int half = lane >> 4;
    const int ln   = lane & 15;
    const int wm = wave & 3;                      // 4 m-strips of 32
    const int wn = wave >> 2;                     // 2 n-strips of 64
    const int mBlk = blockIdx.y * 128;
    const int nBlk = blockIdx.x * 128;

    const v8f vzero = {0.f, 0.f, 0.f, 0.f, 0.f, 0.f, 0.f, 0.f};
    v8f acc[2][4];
#pragma unroll
    for (int mt = 0; mt < 2; mt++)
#pragma unroll
        for (int nt = 0; nt < 4; nt++) acc[mt][nt] = vzero;

    gemm_stage(A, Bt, K, 0, mBlk, nBlk, sA[0], sB[0], tid);

    for (int kb = 0; kb < K; kb += 32) {
        const int cur = (kb >> 5) & 1;
        const bool more = (kb + 32) < K;
        if (more) gemm_stage(A, Bt, K, kb + 32, mBlk, nBlk, sA[cur ^ 1], sB[cur ^ 1], tid);
        if (more) WAIT_ASYNC(4); else WAIT_ASYNC(0);   // 4 async ops/thread per stage
        __syncthreads();

        Frag a0, a1;   // A layout: lane m=ln; e0..7 -> k=half*8+e, e8..15 -> k=16+half*8+e
        const __bf16* ar0 = sA[cur] + (wm * 32 + ln) * 40;
        const __bf16* ar1 = ar0 + 16 * 40;
        a0.u[0] = *(const v4u*)(ar0 + half * 8);
        a0.u[1] = *(const v4u*)(ar0 + 16 + half * 8);
        a1.u[0] = *(const v4u*)(ar1 + half * 8);
        a1.u[1] = *(const v4u*)(ar1 + 16 + half * 8);
#pragma unroll
        for (int nt = 0; nt < 4; nt++) {
            const __bf16* br = sB[cur] + (wn * 64 + nt * 16 + ln) * 40;
            Frag b;
            b.u[0] = *(const v4u*)(br + half * 8);
            b.u[1] = *(const v4u*)(br + 16 + half * 8);
            acc[0][nt] = wmma_bf16(a0, b, acc[0][nt]);
            acc[1][nt] = wmma_bf16(a1, b, acc[1][nt]);
        }
        __syncthreads();
    }

#pragma unroll
    for (int mt = 0; mt < 2; mt++) {
#pragma unroll
        for (int nt = 0; nt < 4; nt++) {
#pragma unroll
            for (int r = 0; r < 8; r++) {
                int m = mBlk + wm * 32 + mt * 16 + r + 8 * half;
                int n = nBlk + wn * 64 + nt * 16 + ln;
                float c = acc[mt][nt][r];
                if (mode == 0) {          // [B,H,S,Dh]
                    outb[(((size_t)((m >> 11) * 16 + (n >> 6)) * 2048 + (m & 2047)) << 6) + (n & 63)] = f2bf(c);
                } else if (mode == 1) {   // [B,H,Dh,S]
                    outb[(((size_t)((m >> 11) * 16 + (n >> 6)) * 64 + (n & 63)) << 11) + (m & 2047)] = f2bf(c);
                } else if (mode == 2) {
                    outf[(size_t)m * N + n] = aux[(size_t)m * N + n] + c;
                } else if (mode == 3) {
                    float z = c + aux[n];
                    float g = 0.5f * z * (1.0f + erff(z * 0.70710678118654752f));
                    outb[(size_t)m * N + n] = f2bf(g);
                } else {
                    outf[(size_t)m * N + n] = c + aux[n] + aux2[(size_t)m * N + n];
                }
            }
        }
    }
}

// ---------------------------------------------------------------------------
// Flash attention with block-cooperative async-to-LDS K/V staging.
// q  : bf16 [B,H,S,Dh]; k : bf16 [B,H,S,Dh]; vt : bf16 [B,H,Dh,S]
// cat: bf16 [B,S,H*Dh]
// Block = 256 thr = 8 waves; 128 query rows; uniform key loop over [0, S0+128).
// ---------------------------------------------------------------------------
__global__ __launch_bounds__(256) void attn_kernel(const __bf16* __restrict__ q,
                                                   const __bf16* __restrict__ k,
                                                   const __bf16* __restrict__ vt,
                                                   __bf16* __restrict__ cat) {
    __shared__ __bf16 sK[2][32 * 72];   // 32 keys x 64 e, row stride 72
    __shared__ __bf16 sV[2][64 * 40];   // 64 e x 32 t,   row stride 40
    __shared__ __bf16 pl[8 * 16 * 40];  // per-wave 16x32 P tile
    const int tid  = threadIdx.x;
    const int lane = tid & 31;
    const int wave = tid >> 5;
    const int half = lane >> 4;
    const int ln   = lane & 15;
    const int bh = blockIdx.y;
    const int S0 = blockIdx.x * 128;
    const int s0 = S0 + wave * 16;

    const __bf16* qbase = q  + (size_t)bh * 2048 * 64;
    const __bf16* kbase = k  + (size_t)bh * 2048 * 64;
    const __bf16* vbase = vt + (size_t)bh * 64 * 2048;

    Frag aq0, aq1;
    {
        const __bf16* qrow = qbase + (size_t)(s0 + ln) * 64;
        aq0.u[0] = *(const v4u*)(qrow + half * 8);
        aq0.u[1] = *(const v4u*)(qrow + 16 + half * 8);
        aq1.u[0] = *(const v4u*)(qrow + 32 + half * 8);
        aq1.u[1] = *(const v4u*)(qrow + 48 + half * 8);
    }

    const v8f vzero = {0.f, 0.f, 0.f, 0.f, 0.f, 0.f, 0.f, 0.f};
    v8f o0 = vzero, o1 = vzero, o2 = vzero, o3 = vzero;
    float rowmax[8], rowsum[8];
#pragma unroll
    for (int r = 0; r < 8; r++) { rowmax[r] = -NEG_BIG; rowsum[r] = 0.f; }

    __bf16* pw = pl + wave * 16 * 40;
    const float scale = 0.125f;          // 1/sqrt(64)
    const int T_end = S0 + 128;

    // stage issuer: 256 chunks K + 256 chunks V -> 1 of each per thread
    auto issue = [&](int t0, int buf) {
        int kr = tid >> 3, ko = (tid & 7) * 8;                 // K: 32 rows x 8 chunks
        async_cp16(kbase + (size_t)(t0 + kr) * 64 + ko, &sK[buf][kr * 72 + ko]);
        int vr = tid >> 2, vo = (tid & 3) * 8;                 // V: 64 rows x 4 chunks
        async_cp16(vbase + (size_t)vr * 2048 + t0 + vo, &sV[buf][vr * 40 + vo]);
    };

    issue(0, 0);
    for (int t0 = 0; t0 < T_end; t0 += 32) {
        const int cur = (t0 >> 5) & 1;
        const bool more = (t0 + 32) < T_end;
        if (more) issue(t0 + 32, cur ^ 1);
        if (more) WAIT_ASYNC(2); else WAIT_ASYNC(0);
        __syncthreads();

        // ---- scores: two 16x16 tiles over keys [t0,t0+16) / [t0+16,t0+32)
        v8f c0 = vzero, c1 = vzero;
        {
            const __bf16* k0 = sK[cur] + ln * 72;
            Frag b0, b1;
            b0.u[0] = *(const v4u*)(k0 + half * 8);
            b0.u[1] = *(const v4u*)(k0 + 16 + half * 8);
            b1.u[0] = *(const v4u*)(k0 + 32 + half * 8);
            b1.u[1] = *(const v4u*)(k0 + 48 + half * 8);
            c0 = wmma_bf16(aq0, b0, c0);
            c0 = wmma_bf16(aq1, b1, c0);
        }
        {
            const __bf16* k1 = sK[cur] + (16 + ln) * 72;
            Frag b0, b1;
            b0.u[0] = *(const v4u*)(k1 + half * 8);
            b0.u[1] = *(const v4u*)(k1 + 16 + half * 8);
            b1.u[0] = *(const v4u*)(k1 + 32 + half * 8);
            b1.u[1] = *(const v4u*)(k1 + 48 + half * 8);
            c1 = wmma_bf16(aq0, b0, c1);
            c1 = wmma_bf16(aq1, b1, c1);
        }
        // ---- causal mask + online softmax
#pragma unroll
        for (int r = 0; r < 8; r++) {
            int srow = s0 + r + 8 * half;
            float x0 = ((t0 + ln)      <= srow) ? c0[r] * scale : -NEG_BIG;
            float x1 = ((t0 + 16 + ln) <= srow) ? c1[r] * scale : -NEG_BIG;
            float tm = fmaxf(x0, x1);
#pragma unroll
            for (int off = 1; off < 16; off <<= 1) tm = fmaxf(tm, __shfl_xor(tm, off, 32));
            float nm = fmaxf(rowmax[r], tm);
            float fct = __expf(rowmax[r] - nm);
            rowmax[r] = nm;
            float p0 = __expf(x0 - nm);
            float p1 = __expf(x1 - nm);
            float ps = p0 + p1;
#pragma unroll
            for (int off = 1; off < 16; off <<= 1) ps += __shfl_xor(ps, off, 32);
            rowsum[r] = rowsum[r] * fct + ps;
            o0[r] *= fct; o1[r] *= fct; o2[r] *= fct; o3[r] *= fct;
            pw[(r + 8 * half) * 40 + ln]      = f2bf(p0);
            pw[(r + 8 * half) * 40 + 16 + ln] = f2bf(p1);
        }
        asm volatile("" ::: "memory");   // DS in-order per wave; pin compiler order
        Frag pa;
        {
            const __bf16* prow = pw + ln * 40;
            pa.u[0] = *(const v4u*)(prow + half * 8);
            pa.u[1] = *(const v4u*)(prow + 16 + half * 8);
        }
        // ---- O += P @ V
#pragma unroll
        for (int t = 0; t < 4; t++) {
            const __bf16* vrow = sV[cur] + (t * 16 + ln) * 40;
            Frag bv;
            bv.u[0] = *(const v4u*)(vrow + half * 8);
            bv.u[1] = *(const v4u*)(vrow + 16 + half * 8);
            v8f& o = (t == 0) ? o0 : (t == 1) ? o1 : (t == 2) ? o2 : o3;
            o = wmma_bf16(pa, bv, o);
        }
        __syncthreads();
    }

    const int b = bh >> 4, h = bh & 15;
#pragma unroll
    for (int r = 0; r < 8; r++) {
        float inv = 1.0f / rowsum[r];
        size_t row = (size_t)(b * 2048 + s0 + r + 8 * half) * 1024 + h * 64;
        cat[row + 0  + ln] = f2bf(o0[r] * inv);
        cat[row + 16 + ln] = f2bf(o1[r] * inv);
        cat[row + 32 + ln] = f2bf(o2[r] * inv);
        cat[row + 48 + ln] = f2bf(o3[r] * inv);
    }
}

// ---------------------------------------------------------------------------
// Host-side orchestration
// ---------------------------------------------------------------------------
extern "C" void kernel_launch(void* const* d_in, const int* in_sizes, int n_in,
                              void* d_out, int out_size, void* d_ws, size_t ws_size,
                              hipStream_t stream) {
    (void)in_sizes; (void)n_in; (void)out_size; (void)ws_size;

    const float* x    = (const float*)d_in[0];
    /* d_in[1] = mask (causal, implicit) */
    const float* Wq   = (const float*)d_in[2];
    const float* Wk   = (const float*)d_in[3];
    const float* Wv   = (const float*)d_in[4];
    const float* Wo   = (const float*)d_in[5];
    const float* ln1w = (const float*)d_in[6];
    const float* ln1b = (const float*)d_in[7];
    const float* ln2w = (const float*)d_in[8];
    const float* ln2b = (const float*)d_in[9];
    const float* W1   = (const float*)d_in[10];
    const float* b1   = (const float*)d_in[11];
    const float* W2   = (const float*)d_in[12];
    const float* b2   = (const float*)d_in[13];
    float* out = (float*)d_out;

    const size_t M = 4096;            // B*S
    char* base = (char*)d_ws;
    size_t off = 0;
    auto alloc = [&](size_t bytes) -> char* {
        char* p = base + off;
        off = (off + bytes + 255) & ~(size_t)255;
        return p;
    };

    __bf16* hb   = (__bf16*)alloc(M * 1024 * 2);
    __bf16* h2b  = (__bf16*)alloc(M * 1024 * 2);
    __bf16* qb   = (__bf16*)alloc(M * 1024 * 2);
    __bf16* kbuf = (__bf16*)alloc(M * 1024 * 2);
    __bf16* vtb  = (__bf16*)alloc(M * 1024 * 2);
    __bf16* cat  = (__bf16*)alloc(M * 1024 * 2);
    float*  attx = (float*) alloc(M * 1024 * 4);
    __bf16* ff1  = (__bf16*)alloc(M * 4096 * 2);
    __bf16* Wqp  = (__bf16*)alloc(1024 * 1024 * 2);
    __bf16* Wkp  = (__bf16*)alloc(1024 * 1024 * 2);
    __bf16* Wvp  = (__bf16*)alloc(1024 * 1024 * 2);
    __bf16* Wop  = (__bf16*)alloc(1024 * 1024 * 2);
    __bf16* W1p  = (__bf16*)alloc(1024 * 4096 * 2);
    __bf16* W2p  = (__bf16*)alloc(4096 * 1024 * 2);

    pack_headed<<<4096, 256, 0, stream>>>(Wq, Wqp);
    pack_headed<<<4096, 256, 0, stream>>>(Wk, Wkp);
    pack_headed<<<4096, 256, 0, stream>>>(Wv, Wvp);
    pack_T<<<4096,  256, 0, stream>>>(Wo, Wop, 1024, 1024);
    pack_T<<<16384, 256, 0, stream>>>(W1, W1p, 4096, 1024);
    pack_T<<<16384, 256, 0, stream>>>(W2, W2p, 1024, 4096);

    ln_kernel<<<4096, 256, 0, stream>>>(x, ln1w, ln1b, hb);

    dim3 g1k(1024 / 128, 4096 / 128);
    gemm_bf16_kernel<<<g1k, 256, 0, stream>>>(hb, Wqp, 1024, 1024, 0, nullptr, nullptr, nullptr, qb);
    gemm_bf16_kernel<<<g1k, 256, 0, stream>>>(hb, Wkp, 1024, 1024, 0, nullptr, nullptr, nullptr, kbuf);
    gemm_bf16_kernel<<<g1k, 256, 0, stream>>>(hb, Wvp, 1024, 1024, 1, nullptr, nullptr, nullptr, vtb);

    attn_kernel<<<dim3(2048 / 128, 32), 256, 0, stream>>>(qb, kbuf, vtb, cat);

    gemm_bf16_kernel<<<g1k, 256, 0, stream>>>(cat, Wop, 1024, 1024, 2, x, nullptr, attx, nullptr);

    ln_kernel<<<4096, 256, 0, stream>>>(attx, ln2w, ln2b, h2b);

    dim3 g4k(4096 / 128, 4096 / 128);
    gemm_bf16_kernel<<<g4k, 256, 0, stream>>>(h2b, W1p, 4096, 1024, 3, b1, nullptr, nullptr, ff1);
    gemm_bf16_kernel<<<g1k, 256, 0, stream>>>(ff1, W2p, 1024, 4096, 4, b2, attx, out, nullptr);
}
